// PerceiverResampler_68942815036088
// MI455X (gfx1250) — matrix operand order repaired and checked by reading, not compiled, not run on hardware
//
#include <hip/hip_runtime.h>
#include <math.h>
#include <stdint.h>

typedef __attribute__((ext_vector_type(16))) __bf16 v16bf;
typedef __attribute__((ext_vector_type(8)))  float  v8f;

#define BM 128
#define BN 128
#define BK 64
#define SA 72                    // padded K-major stride (bf16 elems); 144B = 9*16B
#define TILEB (BM * SA * 2)      // bytes per LDS tile buffer (18432)
#define BUFU4 (TILEB / 16)       // uint4 elements between double buffers (1152)

__device__ __forceinline__ unsigned short f2bf(float f) {
  unsigned int u = __builtin_bit_cast(unsigned int, f);
  unsigned int r = u + 0x7FFFu + ((u >> 16) & 1u);   // round-to-nearest-even
  return (unsigned short)(r >> 16);
}
__device__ __forceinline__ unsigned f2bf2(float a, float b) {
  return (unsigned)f2bf(a) | ((unsigned)f2bf(b) << 16);
}

// ---------------------------------------------------------------------------
// GEMM: C[orow,N] = epilogue( A[M,K](bf16) @ Wt[N,K](bf16)^T )
// Both LDS tiles filled with GLOBAL_LOAD_ASYNC_TO_LDS_B128 (ASYNCcnt),
// double-buffered with compile-time buffer selection; fragments read as
// explicit uint4 (ds_load_b128) pairs.
// ---------------------------------------------------------------------------
template<bool BIAS, bool RES, bool GELU, bool OUTBF>
__global__ __launch_bounds__(256)
void gemm_wmma(const unsigned short* __restrict__ A,
               const unsigned short* __restrict__ Wt,
               const float* __restrict__ bias, const float* __restrict__ bias2,
               const float* __restrict__ res, void* __restrict__ Cout,
               int N, int K,
               unsigned rpb_in, unsigned rpb_out, unsigned row_off, float scale)
{
  __shared__ __align__(16) unsigned short Asb[2 * BM * SA];  // [row][k]
  __shared__ __align__(16) unsigned short Bsb[2 * BN * SA];  // [col][k]

  const int tid  = threadIdx.x;
  const int lane = tid & 31;
  const int wid  = tid >> 5;
  const int wm   = wid >> 2;         // 0..1  (64-row slab)
  const int wn   = wid & 3;          // 0..3  (32-col slab)
  const int l16  = lane & 15;
  const int half = lane >> 4;
  const int bm0  = blockIdx.y * BM;
  const int bn0  = blockIdx.x * BN;

  const unsigned asbFlat = (unsigned)(uintptr_t)(&Asb[0]);
  const unsigned bsbFlat = (unsigned)(uintptr_t)(&Bsb[0]);

  v8f acc[4][2];
  #pragma unroll
  for (int a = 0; a < 4; ++a)
    #pragma unroll
    for (int b = 0; b < 2; ++b)
      #pragma unroll
      for (int e = 0; e < 8; ++e) acc[a][b][e] = 0.0f;

  // hoisted fragment base pointers (buffer 0); buffer 1 = +BUFU4 uint4 elems
  const uint4* pA[4][2];
  const uint4* pB[2][2];
  #pragma unroll
  for (int tm = 0; tm < 4; ++tm)
    #pragma unroll
    for (int ks = 0; ks < 2; ++ks)
      pA[tm][ks] = (const uint4*)((const char*)Asb +
                   (wm * 64 + tm * 16 + l16) * (SA * 2) + ks * 64 + 16 * half);
  #pragma unroll
  for (int tn = 0; tn < 2; ++tn)
    #pragma unroll
    for (int ks = 0; ks < 2; ++ks)
      pB[tn][ks] = (const uint4*)((const char*)Bsb +
                   (wn * 32 + tn * 16 + l16) * (SA * 2) + ks * 64 + 16 * half);

  // async-stage one 128x64 K-slab of A and B into buffer at byte offset bufb
  auto issue = [&](int k0, unsigned bufb) {
    #pragma unroll
    for (int i = 0; i < 4; ++i) {
      int c = tid + i * 256;                       // A: 128 rows x 8 x 16B
      int r = c >> 3, q8 = c & 7;
      unsigned long long ga =
          (unsigned long long)(uintptr_t)(A + (size_t)(bm0 + r) * K + k0 + q8 * 8);
      unsigned la = asbFlat + bufb + (unsigned)(r * (SA * 2) + q8 * 16);
      asm volatile("global_load_async_to_lds_b128 %0, %1, off"
                   :: "v"(la), "v"(ga) : "memory");
    }
    #pragma unroll
    for (int i = 0; i < 4; ++i) {
      int c = tid + i * 256;                       // B: 128 cols x 8 x 16B
      int r = c >> 3, q8 = c & 7;
      unsigned long long ga =
          (unsigned long long)(uintptr_t)(Wt + (size_t)(bn0 + r) * K + k0 + q8 * 8);
      unsigned la = bsbFlat + bufb + (unsigned)(r * (SA * 2) + q8 * 16);
      asm volatile("global_load_async_to_lds_b128 %0, %1, off"
                   :: "v"(la), "v"(ga) : "memory");
    }
  };

  // 16 WMMAs over one staged 64-deep slab; bufofs is a compile-time constant
  auto compute = [&](int bufofs) {
    union Frag { v16bf v; uint4 q[2]; };
    #pragma unroll
    for (int ks = 0; ks < 2; ++ks) {
      Frag bF[2];
      #pragma unroll
      for (int tn = 0; tn < 2; ++tn) {
        bF[tn].q[0] = pB[tn][ks][bufofs];
        bF[tn].q[1] = pB[tn][ks][bufofs + 2];
      }
      #pragma unroll
      for (int tm = 0; tm < 4; ++tm) {
        Frag aF;
        aF.q[0] = pA[tm][ks][bufofs];
        aF.q[1] = pA[tm][ks][bufofs + 2];          // +32 bytes
        #pragma unroll
        for (int tn = 0; tn < 2; ++tn)
          acc[tm][tn] = __builtin_amdgcn_wmma_f32_16x16x32_bf16(
              false, aF.v, false, bF[tn].v, (short)0, acc[tm][tn], false, false);
      }
    }
  };

  // software pipeline, K % 128 == 0 for all call sites
  issue(0, 0);
  asm volatile("s_wait_asynccnt 0" ::: "memory");
  __syncthreads();
  for (int k0 = 0; k0 < K; k0 += 2 * BK) {
    issue(k0 + BK, TILEB);                 // k0+64 <= K-64 always
    compute(0);
    asm volatile("s_wait_asynccnt 0" ::: "memory");
    __syncthreads();
    if (k0 + 2 * BK < K) issue(k0 + 2 * BK, 0);
    compute(BUFU4);
    if (k0 + 2 * BK < K) asm volatile("s_wait_asynccnt 0" ::: "memory");
    __syncthreads();
  }

  // epilogue: C/D layout row = v + 8*half, col = l16
  #pragma unroll
  for (int tm = 0; tm < 4; ++tm) {
    #pragma unroll
    for (int tn = 0; tn < 2; ++tn) {
      int col = bn0 + wn * 32 + tn * 16 + l16;
      float bv = 0.0f;
      if (BIAS) { bv = bias[col]; if (bias2) bv += bias2[col]; }
      #pragma unroll
      for (int v = 0; v < 8; ++v) {
        unsigned row  = bm0 + wm * 64 + tm * 16 + v + half * 8;
        unsigned orow = (rpb_in == rpb_out)
                          ? row + row_off
                          : (row / rpb_in) * rpb_out + (row % rpb_in) + row_off;
        float val = acc[tm][tn][v] * scale;
        if (BIAS) val += bv;
        size_t oi = (size_t)orow * N + col;
        if (RES) val += res[oi];
        if (GELU) val = 0.5f * val * (1.0f + erff(val * 0.70710678118654752f));
        if (OUTBF) ((unsigned short*)Cout)[oi] = f2bf(val);
        else       ((float*)Cout)[oi] = val;
      }
    }
  }
}

// ---------------------------------------------------------------------------
// weight transform: W[K,N] f32 -> Wt[N,K] bf16 (tiled transpose + convert)
// ---------------------------------------------------------------------------
__global__ __launch_bounds__(256)
void transpose_bf16_k(const float* __restrict__ W, unsigned short* __restrict__ Wt,
                      int K, int N)
{
  __shared__ float tile[32][33];
  const int lx = threadIdx.x & 31;
  const int ly = threadIdx.x >> 5;
  const int bx = blockIdx.x;   // N tile
  const int by = blockIdx.y;   // K tile
  #pragma unroll
  for (int i = 0; i < 4; ++i) {
    int k = by * 32 + ly + i * 8;
    tile[ly + i * 8][lx] = W[(size_t)k * N + bx * 32 + lx];
  }
  __syncthreads();
  #pragma unroll
  for (int i = 0; i < 4; ++i) {
    int n = bx * 32 + ly + i * 8;
    Wt[(size_t)n * K + by * 32 + lx] = f2bf(tile[lx][ly + i * 8]);
  }
}

// elementwise f32 -> bf16 (image features), float4 granularity
__global__ void cvtbf_k(const float* __restrict__ s, unsigned short* __restrict__ d,
                        long n4)
{
  long i = (long)blockIdx.x * 256 + threadIdx.x;
  if (i < n4) {
    float4 v = ((const float4*)s)[i];
    uint2 u; u.x = f2bf2(v.x, v.y); u.y = f2bf2(v.z, v.w);
    ((uint2*)d)[i] = u;
  }
}

// rowwise layernorm, D=1024, one block/row, bf16 output (feeds GEMM A operand)
__global__ __launch_bounds__(256)
void layernorm_k(const float* __restrict__ x, const float* __restrict__ g,
                 const float* __restrict__ b, unsigned short* __restrict__ y)
{
  __shared__ float rs[8], rss[8], stats[2];
  const int tid = threadIdx.x, lane = tid & 31, wid = tid >> 5;
  const size_t row = blockIdx.x;
  const float4 t = *(const float4*)(x + row * 1024 + tid * 4);
  float s  = t.x + t.y + t.z + t.w;
  float ss = t.x*t.x + t.y*t.y + t.z*t.z + t.w*t.w;
  #pragma unroll
  for (int o = 16; o > 0; o >>= 1) { s += __shfl_down(s, o); ss += __shfl_down(ss, o); }
  if (lane == 0) { rs[wid] = s; rss[wid] = ss; }
  __syncthreads();
  if (tid == 0) {
    float S = 0.0f, SS = 0.0f;
    #pragma unroll
    for (int w = 0; w < 8; ++w) { S += rs[w]; SS += rss[w]; }
    float mean = S * (1.0f / 1024.0f);
    float var  = SS * (1.0f / 1024.0f) - mean * mean;
    stats[0] = mean; stats[1] = rsqrtf(var + 1e-5f);
  }
  __syncthreads();
  const float mean = stats[0], rstd = stats[1];
  const float4 gg = *(const float4*)(g + tid * 4);
  const float4 bb = *(const float4*)(b + tid * 4);
  uint2 u;
  u.x = f2bf2((t.x - mean) * rstd * gg.x + bb.x, (t.y - mean) * rstd * gg.y + bb.y);
  u.y = f2bf2((t.z - mean) * rstd * gg.z + bb.z, (t.w - mean) * rstd * gg.w + bb.w);
  *(uint2*)(y + row * 1024 + tid * 4) = u;
}

// attention for one (b,h,latent-row) per block; q pre-scaled; bf16 output
__global__ __launch_bounds__(256)
void attention_k(const float* __restrict__ q, const float* __restrict__ kv,
                 unsigned short* __restrict__ out)
{
  __shared__ float qs[64];
  __shared__ float sim[640];
  __shared__ float red[8];
  __shared__ float part[4][64];

  const int tid = threadIdx.x;
  const int lane = tid & 31, wid = tid >> 5;
  const int i  = blockIdx.x;
  const int bh = blockIdx.y;
  const int b  = bh >> 4;
  const int h  = bh & 15;

  const float* qrow = q + ((size_t)(b * 64 + i)) * 1024 + h * 64;
  if (tid < 64) qs[tid] = qrow[tid];
  __syncthreads();

  const float* kbase = kv + (size_t)b * 640 * 2048 + h * 64;
  float lmax = -3.0e38f;
  for (int j = tid; j < 640; j += 256) {
    const float* kr = kbase + (size_t)j * 2048;
    float d = 0.0f;
    #pragma unroll
    for (int t = 0; t < 64; ++t) d = fmaf(qs[t], kr[t], d);
    sim[j] = d;
    lmax = fmaxf(lmax, d);
  }
  #pragma unroll
  for (int o = 16; o > 0; o >>= 1) lmax = fmaxf(lmax, __shfl_xor(lmax, o));
  if (lane == 0) red[wid] = lmax;
  __syncthreads();
  float m = red[0];
  #pragma unroll
  for (int w = 1; w < 8; ++w) m = fmaxf(m, red[w]);

  float lsum = 0.0f;
  for (int j = tid; j < 640; j += 256) {
    float p = expf(sim[j] - m);
    sim[j] = p;
    lsum += p;
  }
  #pragma unroll
  for (int o = 16; o > 0; o >>= 1) lsum += __shfl_xor(lsum, o);
  __syncthreads();
  if (lane == 0) red[wid] = lsum;
  __syncthreads();
  float denom = 0.0f;
  #pragma unroll
  for (int w = 0; w < 8; ++w) denom += red[w];
  const float inv = 1.0f / denom;

  const float* vbase = kbase + 1024;
  const int d  = tid & 63;
  const int gg = tid >> 6;
  float a = 0.0f;
  for (int j = gg; j < 640; j += 4)
    a = fmaf(sim[j], vbase[(size_t)j * 2048 + d], a);
  part[gg][d] = a;
  __syncthreads();
  if (tid < 64) {
    float s = part[0][tid] + part[1][tid] + part[2][tid] + part[3][tid];
    out[((size_t)(b * 64 + i)) * 1024 + h * 64 + tid] = f2bf(s * inv);
  }
}

__global__ void bcast_lat(const float* __restrict__ src, float* __restrict__ dst) {
  size_t i = (size_t)blockIdx.x * blockDim.x + threadIdx.x;
  if (i < (size_t)32 * 64 * 1024) dst[i] = src[i & 65535];
}

extern "C" void kernel_launch(void* const* d_in, const int* in_sizes, int n_in,
                              void* d_out, int out_size, void* d_ws, size_t ws_size,
                              hipStream_t stream)
{
  (void)in_sizes; (void)n_in; (void)out_size; (void)ws_size;
  int ii = 0;
  auto nx = [&]() { return (const float*)d_in[ii++]; };

  const float* image = nx();
  const float* lat0  = nx();
  const float* pos   = nx();
  const float* vis_W = nx();
  const float* vis_b = nx();
  struct Layer { const float *ln_m_g, *ln_m_b, *ln_l_g, *ln_l_b,
                             *Wq, *Wkv, *Wo, *ln_f_g, *ln_f_b, *W1, *W2; } Ls[6];
  for (int l = 0; l < 6; ++l) {
    Ls[l].ln_m_g = nx(); Ls[l].ln_m_b = nx();
    Ls[l].ln_l_g = nx(); Ls[l].ln_l_b = nx();
    Ls[l].Wq = nx(); Ls[l].Wkv = nx(); Ls[l].Wo = nx();
    Ls[l].ln_f_g = nx(); Ls[l].ln_f_b = nx();
    Ls[l].W1 = nx(); Ls[l].W2 = nx();
  }
  const float* norm_g = nx();
  const float* norm_b = nx();
  const float* llm_W  = nx();
  const float* llm_b  = nx();

  // ---- workspace carve -----------------------------------------------------
  float* ws = (float*)d_ws;
  size_t o = 0;
  float* x   = ws + o; o += (size_t)32 * 576 * 1024;   // vis projection (f32)
  float* lat = ws + o; o += (size_t)32 * 64 * 1024;    // residual stream
  float* q   = ws + o; o += (size_t)32 * 64 * 1024;
  float* kv  = ws + o; o += (size_t)32 * 640 * 2048;

  unsigned short* u16 = (unsigned short*)(ws + o);
  size_t uo = 0;
  auto carve = [&](size_t n) { unsigned short* p = u16 + uo; uo += n; return p; };
  unsigned short* imgb = carve((size_t)32 * 576 * 1024);  // image bf16; reused as xm
  unsigned short* xmb  = imgb;                            // alias (image used once)
  unsigned short* lmb  = carve((size_t)32 * 64 * 1024);
  unsigned short* aob  = carve((size_t)32 * 64 * 1024);
  unsigned short* hbb  = carve((size_t)32 * 64 * 4096);
  unsigned short* visWt = carve((size_t)1024 * 1024);
  unsigned short* llmWt = carve((size_t)1024 * 4096);
  unsigned short *WqT[6], *WkvT[6], *WoT[6], *W1T[6], *W2T[6];
  for (int l = 0; l < 6; ++l) {
    WqT[l]  = carve((size_t)1024 * 1024);
    WkvT[l] = carve((size_t)1024 * 2048);
    WoT[l]  = carve((size_t)1024 * 1024);
    W1T[l]  = carve((size_t)1024 * 4096);
    W2T[l]  = carve((size_t)4096 * 1024);
  }

  const int MX = 32 * 576;   // 18432 media rows
  const int ML = 32 * 64;    // 2048 latent rows
  dim3 blk(256);
  auto G = [](int Mr, int Nc) { return dim3(Nc / 128, Mr / 128); };
  auto tr = [&](const float* Wsrc, unsigned short* Wdst, int K_, int N_) {
    transpose_bf16_k<<<dim3(N_ / 32, K_ / 32), 256, 0, stream>>>(Wsrc, Wdst, K_, N_);
  };

  // ---- one-time-per-launch transforms -------------------------------------
  cvtbf_k<<<(MX * 1024 / 4 + 255) / 256, 256, 0, stream>>>(image, imgb,
                                                           (long)MX * 1024 / 4);
  tr(vis_W, visWt, 1024, 1024);
  tr(llm_W, llmWt, 1024, 4096);
  for (int l = 0; l < 6; ++l) {
    tr(Ls[l].Wq,  WqT[l],  1024, 1024);
    tr(Ls[l].Wkv, WkvT[l], 1024, 2048);
    tr(Ls[l].Wo,  WoT[l],  1024, 1024);
    tr(Ls[l].W1,  W1T[l],  1024, 4096);
    tr(Ls[l].W2,  W2T[l],  4096, 1024);
  }
  bcast_lat<<<(32 * 64 * 1024 + 255) / 256, 256, 0, stream>>>(lat0, lat);

  // x = image @ vis_W + vis_b + pos   (pos folded as second bias vector)
  gemm_wmma<true, false, false, false><<<G(MX, 1024), blk, 0, stream>>>(
      imgb, visWt, vis_b, pos, nullptr, x, 1024, 1024, 1u, 1u, 0u, 1.0f);

  for (int l = 0; l < 6; ++l) {
    const Layer& P = Ls[l];
    layernorm_k<<<MX, 256, 0, stream>>>(x,   P.ln_m_g, P.ln_m_b, xmb);
    layernorm_k<<<ML, 256, 0, stream>>>(lat, P.ln_l_g, P.ln_l_b, lmb);
    // q = (ln(lat) @ Wq) * DH^-0.5
    gemm_wmma<false, false, false, false><<<G(ML, 1024), blk, 0, stream>>>(
        lmb, WqT[l], nullptr, nullptr, nullptr, q, 1024, 1024, 1u, 1u, 0u, 0.125f);
    // kv = concat(xm, lm) @ Wkv  — row-remapped into [B,640,2048]
    gemm_wmma<false, false, false, false><<<G(MX, 2048), blk, 0, stream>>>(
        xmb, WkvT[l], nullptr, nullptr, nullptr, kv, 2048, 1024, 576u, 640u, 0u, 1.0f);
    gemm_wmma<false, false, false, false><<<G(ML, 2048), blk, 0, stream>>>(
        lmb, WkvT[l], nullptr, nullptr, nullptr, kv, 2048, 1024, 64u, 640u, 576u, 1.0f);
    attention_k<<<dim3(64, 512), 256, 0, stream>>>(q, kv, aob);
    // lat = ao @ Wo + lat
    gemm_wmma<false, true, false, false><<<G(ML, 1024), blk, 0, stream>>>(
        aob, WoT[l], nullptr, nullptr, lat, lat, 1024, 1024, 1u, 1u, 0u, 1.0f);
    layernorm_k<<<ML, 256, 0, stream>>>(lat, P.ln_f_g, P.ln_f_b, lmb);
    // hb = gelu(ln(lat) @ W1)  (bf16 out, feeds next GEMM)
    gemm_wmma<false, false, true, true><<<G(ML, 4096), blk, 0, stream>>>(
        lmb, W1T[l], nullptr, nullptr, nullptr, hbb, 4096, 1024, 1u, 1u, 0u, 1.0f);
    // lat = hb @ W2 + lat
    gemm_wmma<false, true, false, false><<<G(ML, 1024), blk, 0, stream>>>(
        hbb, W2T[l], nullptr, nullptr, lat, lat, 1024, 4096, 1u, 1u, 0u, 1.0f);
  }

  layernorm_k<<<ML, 256, 0, stream>>>(lat, norm_g, norm_b, lmb);
  gemm_wmma<true, false, false, false><<<G(ML, 4096), blk, 0, stream>>>(
      lmb, llmWt, llm_b, nullptr, nullptr, (float*)d_out, 4096, 1024, 1u, 1u, 0u, 1.0f);
}
